// LSTM_4578435137888
// MI455X (gfx1250) — matrix operand order repaired and checked by reading, not compiled, not run on hardware
//
#include <hip/hip_runtime.h>
#include <hip/hip_bf16.h>

// ---------------------------------------------------------------------------
// LSTM  B=64 T=512 I=256 H=1024 O=128   (gate order i,f,g,o)
//   K1: convert x / W_ih / W_hh to bf16, bias = b_ih + b_hh
//   K2: x_proj = x @ W_ih^T  (bf16 WMMA, f32 accum) -> bf16 [32768 x 4096]
//   K3: persistent recurrence, 64 WGs x 4 waves:
//         - W_hh slice (64 gate cols) resident in LDS
//         - h staged into LDS each step via global_load_async_to_lds_b128
//         - double-buffered ds_load_b128 fragment pipeline feeding WMMA
//         - x_proj gate biases prefetched at step top (hidden behind GEMM)
//         - c-state in registers, packed b128 h write-back, grid barrier
//   K4: out = h_T @ W_ho^T + b_ho
// ---------------------------------------------------------------------------

typedef __attribute__((ext_vector_type(16))) __bf16 v16bf;
typedef __attribute__((ext_vector_type(8)))  float  v8f;

union Frag16 { v16bf v; uint4 q[2]; };
union Pack8  { uint4 q; unsigned short s[8]; };

static __device__ __forceinline__ unsigned short f2bf(float x) {
    union { float f; unsigned u; } v; v.f = x;
    unsigned r = v.u + 0x7fffu + ((v.u >> 16) & 1u);   // RNE
    return (unsigned short)(r >> 16);
}
static __device__ __forceinline__ float bf2f(unsigned short h) {
    union { unsigned u; float f; } v; v.u = ((unsigned)h) << 16;
    return v.f;
}
static __device__ __forceinline__ float sigmoidf(float x) {
    return 1.0f / (1.0f + __expf(-x));
}

#define BT      32768     // B*T rows of x_proj
#define HH      1024
#define G4      4096
#define NWG     64        // persistent workgroups (each owns 16 h-columns)
#define WSTRIDE 1032      // LDS row stride in ushorts (1024 + 8 pad -> 2064 B)
// LDS: W slice [64][1032]u16 | h stage [64][1032]u16 | gates [4][64][16]f32
#define LDS_W   0
#define LDS_H   (64 * WSTRIDE * 2)
#define LDS_G   (2 * 64 * WSTRIDE * 2)
#define SMEM_BYTES (LDS_G + 4 * 64 * 16 * 4)     // 280576

// -------------------------------- converts ---------------------------------
__global__ void k_f32_to_bf16(const float* __restrict__ src,
                              unsigned short* __restrict__ dst, int n) {
    int i = blockIdx.x * blockDim.x + threadIdx.x;
    if (i < n) dst[i] = f2bf(src[i]);
}
__global__ void k_bias(const float* __restrict__ bih, const float* __restrict__ bhh,
                       float* __restrict__ bias) {
    int i = blockIdx.x * blockDim.x + threadIdx.x;
    if (i < G4) bias[i] = bih[i] + bhh[i];
}

// ------------------------------ x_proj GEMM --------------------------------
// block = 256 thr (8 waves). Each wave: 64 rows x 16 cols, K=256.
// grid = (4096/16, 32768/512). Double-buffered fragment pipeline.
__global__ void k_xproj(const unsigned short* __restrict__ xbf,   // [32768][256]
                        const unsigned short* __restrict__ wbf,   // [4096][256]
                        const float* __restrict__ bias,           // [4096]
                        unsigned short* __restrict__ xproj)       // [32768][4096]
{
    const int lane = threadIdx.x & 31;
    const int wave = threadIdx.x >> 5;
    const int hi   = lane >> 4;
    const int ln   = lane & 15;
    const int n0   = blockIdx.x * 16;
    const int m0   = blockIdx.y * 512 + wave * 64;

    const unsigned short* bbase = wbf + (size_t)(n0 + ln) * 256 + hi * 8;
    const unsigned short* abase = xbf + (size_t)(m0 + ln) * 256 + hi * 8;

    v8f acc[4] = {};
    Frag16 bfr[2], afr[2][4];

    // prologue: k-step 0 into buffer 0
    bfr[0].q[0] = *(const uint4*)(bbase);
    bfr[0].q[1] = *(const uint4*)(bbase + 16);
#pragma unroll
    for (int mt = 0; mt < 4; ++mt) {
        const unsigned short* ap = abase + (size_t)mt * 16 * 256;
        afr[0][mt].q[0] = *(const uint4*)(ap);
        afr[0][mt].q[1] = *(const uint4*)(ap + 16);
    }

#pragma unroll
    for (int ks = 0; ks < 8; ++ks) {
        const int cur = ks & 1, nxt = cur ^ 1;
        if (ks < 7) {               // prefetch next k-step into other buffer
            const int kk = (ks + 1) * 32;
            bfr[nxt].q[0] = *(const uint4*)(bbase + kk);
            bfr[nxt].q[1] = *(const uint4*)(bbase + kk + 16);
#pragma unroll
            for (int mt = 0; mt < 4; ++mt) {
                const unsigned short* ap = abase + (size_t)mt * 16 * 256 + kk;
                afr[nxt][mt].q[0] = *(const uint4*)(ap);
                afr[nxt][mt].q[1] = *(const uint4*)(ap + 16);
            }
        }
#pragma unroll
        for (int mt = 0; mt < 4; ++mt)
            acc[mt] = __builtin_amdgcn_wmma_f32_16x16x32_bf16(
                false, afr[cur][mt].v, false, bfr[cur].v, (short)0, acc[mt],
                false, false);
    }

    const float bn = bias[n0 + ln];
#pragma unroll
    for (int mt = 0; mt < 4; ++mt)
#pragma unroll
        for (int r = 0; r < 8; ++r) {
            int m = m0 + mt * 16 + hi * 8 + r;
            xproj[(size_t)m * G4 + n0 + ln] = f2bf(acc[mt][r] + bn);
        }
}

// --------------------------- persistent recurrence -------------------------
// 64 blocks x 128 thr (4 waves). Block wg owns h-cols [wg*16, wg*16+16).
// Wave w computes gate strip w (i/f/g/o) for those 16 columns, M=64, K=1024.
__global__ void k_lstm(const unsigned short* __restrict__ whh_bf, // [4096][1024]
                       const unsigned short* __restrict__ xproj,  // [32768][4096]
                       unsigned short* __restrict__ hbuf,         // [2][64][1024]
                       unsigned* __restrict__ sync_cnt)
{
    extern __shared__ unsigned char smem[];
    unsigned short* wlds = (unsigned short*)(smem + LDS_W);   // [64][WSTRIDE]
    unsigned short* hlds = (unsigned short*)(smem + LDS_H);   // [64][WSTRIDE]
    float*          gbuf = (float*)(smem + LDS_G);            // [4][64][16]

    const int tid  = threadIdx.x;          // 0..127
    const int lane = tid & 31;
    const int wave = tid >> 5;             // 0..3 == gate strip (i,f,g,o)
    const int hi   = lane >> 4;
    const int ln   = lane & 15;
    const int wg   = blockIdx.x;

    // ---- stage W_hh slice into LDS once (64 gate rows, padded) ----
    {
        int lc   = tid >> 1;               // local gate row 0..63
        int half = tid & 1;                // 512 ushorts each
        int gcol = (lc >> 4) * HH + wg * 16 + (lc & 15);
        const uint4* src = (const uint4*)(whh_bf + (size_t)gcol * HH + half * 512);
        uint4* dst = (uint4*)(wlds + lc * WSTRIDE + half * 512);
#pragma unroll 8
        for (int i = 0; i < 32; ++i) dst[i] = src[i];
    }

    // ---- c-state in registers: thread -> batch eb, h-cols ejl..ejl+7 ----
    const int eb  = tid >> 1;              // 0..63
    const int ejl = (tid & 1) * 8;         // 0 or 8
    float creg[8];
#pragma unroll
    for (int i = 0; i < 8; ++i) creg[i] = 0.0f;

    // per-thread bases for the per-step async h staging (512 B each)
    const int srow  = tid >> 1;            // 0..63
    const int shalf = tid & 1;
    const unsigned hloff = (unsigned)(uintptr_t)(hlds + srow * WSTRIDE + shalf * 512);

    const unsigned short* brow = wlds + (wave * 16 + ln) * WSTRIDE + hi * 8;
    const unsigned short* arow = hlds + (size_t)ln * WSTRIDE + hi * 8;

    const size_t xcol = (size_t)wg * 16 + ejl;       // this thread's gate cols
    const size_t hrow = (size_t)eb * HH + wg * 16 + ejl;

    __syncthreads();

    for (int t = 0; t < 512; ++t) {
        const unsigned short* hin  = hbuf + (size_t)(t & 1) * (64 * HH);
        unsigned short*       hout = hbuf + (size_t)((t + 1) & 1) * (64 * HH);

        // ---- prefetch x_proj gate biases for this step (8 bf16 per gate) ----
        const unsigned short* xp = xproj + (size_t)(eb * 512 + t) * G4 + xcol;
        Pack8 xpi, xpf, xpg, xpo;
        xpi.q = *(const uint4*)(xp);
        xpf.q = *(const uint4*)(xp + 1024);
        xpg.q = *(const uint4*)(xp + 2048);
        xpo.q = *(const uint4*)(xp + 3072);

        // ---- async-stage h[64][1024] into LDS (CDNA5 async-to-LDS path) ----
        {
            const unsigned short* gsrc = hin + (size_t)srow * HH + shalf * 512;
#pragma unroll
            for (int i = 0; i < 32; ++i)
                asm volatile("global_load_async_to_lds_b128 %0, %1, off offset:%2"
                             :: "v"(hloff), "v"(gsrc), "i"(i * 16));
            asm volatile("s_wait_asynccnt 0x0" ::: "memory");
        }
        __syncthreads();

        // ---- gates_strip = h @ W_strip^T  (M=64, N=16/wave, K=1024) ----
        v8f acc[4] = {};
        Frag16 bfr[2], afr[2][4];
        bfr[0].q[0] = *(const uint4*)(brow);
        bfr[0].q[1] = *(const uint4*)(brow + 16);
#pragma unroll
        for (int mt = 0; mt < 4; ++mt) {
            const unsigned short* ap = arow + (size_t)mt * 16 * WSTRIDE;
            afr[0][mt].q[0] = *(const uint4*)(ap);
            afr[0][mt].q[1] = *(const uint4*)(ap + 16);
        }
#pragma unroll
        for (int ks = 0; ks < 32; ++ks) {
            const int cur = ks & 1, nxt = cur ^ 1;
            if (ks < 31) {          // prefetch next k-step into other buffer
                const int kk = (ks + 1) * 32;
                bfr[nxt].q[0] = *(const uint4*)(brow + kk);
                bfr[nxt].q[1] = *(const uint4*)(brow + kk + 16);
#pragma unroll
                for (int mt = 0; mt < 4; ++mt) {
                    const unsigned short* ap = arow + (size_t)mt * 16 * WSTRIDE + kk;
                    afr[nxt][mt].q[0] = *(const uint4*)(ap);
                    afr[nxt][mt].q[1] = *(const uint4*)(ap + 16);
                }
            }
#pragma unroll
            for (int mt = 0; mt < 4; ++mt)
                acc[mt] = __builtin_amdgcn_wmma_f32_16x16x32_bf16(
                    false, afr[cur][mt].v, false, bfr[cur].v, (short)0, acc[mt],
                    false, false);
        }

        // ---- publish gate strip to LDS ----
#pragma unroll
        for (int mt = 0; mt < 4; ++mt)
#pragma unroll
            for (int r = 0; r < 8; ++r) {
                int b = mt * 16 + hi * 8 + r;
                gbuf[(wave * 64 + b) * 16 + ln] = acc[mt][r];
            }
        __syncthreads();

        // ---- fused elementwise cell update (8 elements / thread) ----
        float gi[8], gf[8], gg[8], go[8];
        {
            const float* gb = gbuf + eb * 16 + ejl;
            *(float4*)&gi[0] = *(const float4*)(gb + 0 * 1024);
            *(float4*)&gi[4] = *(const float4*)(gb + 0 * 1024 + 4);
            *(float4*)&gf[0] = *(const float4*)(gb + 1 * 1024);
            *(float4*)&gf[4] = *(const float4*)(gb + 1 * 1024 + 4);
            *(float4*)&gg[0] = *(const float4*)(gb + 2 * 1024);
            *(float4*)&gg[4] = *(const float4*)(gb + 2 * 1024 + 4);
            *(float4*)&go[0] = *(const float4*)(gb + 3 * 1024);
            *(float4*)&go[4] = *(const float4*)(gb + 3 * 1024 + 4);
        }
        Pack8 ho;
#pragma unroll
        for (int i = 0; i < 8; ++i) {
            float vi = gi[i] + bf2f(xpi.s[i]);
            float vf = gf[i] + bf2f(xpf.s[i]);
            float vg = gg[i] + bf2f(xpg.s[i]);
            float vo = go[i] + bf2f(xpo.s[i]);
            float c  = sigmoidf(vf) * creg[i] + sigmoidf(vi) * tanhf(vg);
            creg[i] = c;
            ho.s[i] = f2bf(sigmoidf(vo) * tanhf(c));
        }
        *(uint4*)(hout + hrow) = ho.q;     // packed b128 h write-back

        __threadfence();        // release new h to device scope
        __syncthreads();        // gbuf/hlds safe for reuse

        // ---- grid barrier (monotonic counter) ----
        if (tid == 0) {
            atomicAdd(sync_cnt, 1u);
            unsigned target = (unsigned)(t + 1) * NWG;
            while (__hip_atomic_load(sync_cnt, __ATOMIC_RELAXED,
                                     __HIP_MEMORY_SCOPE_AGENT) < target)
                __builtin_amdgcn_s_sleep(1);
        }
        __syncthreads();
        __threadfence();        // acquire other WGs' h writes
    }
}

// ------------------------------ final projection ---------------------------
__global__ void k_final(const unsigned short* __restrict__ hfin, // [64][1024] bf16
                        const float* __restrict__ who,           // [128][1024]
                        const float* __restrict__ bho,           // [128]
                        float* __restrict__ out)                 // [64][128]
{
    int idx = blockIdx.x * blockDim.x + threadIdx.x;   // 8192
    if (idx >= 64 * 128) return;
    int b = idx >> 7, o = idx & 127;
    const float* w = who + (size_t)o * HH;
    const unsigned short* h = hfin + (size_t)b * HH;
    float s = bho[o];
    for (int k = 0; k < HH; ++k) s += bf2f(h[k]) * w[k];
    out[idx] = s;
}

// --------------------------------- launch ----------------------------------
extern "C" void kernel_launch(void* const* d_in, const int* in_sizes, int n_in,
                              void* d_out, int out_size, void* d_ws, size_t ws_size,
                              hipStream_t stream) {
    const float* x    = (const float*)d_in[0];   // [64,512,256]
    const float* Wih  = (const float*)d_in[1];   // [4096,256]
    const float* Whh  = (const float*)d_in[2];   // [4096,1024]
    const float* bih  = (const float*)d_in[3];
    const float* bhh  = (const float*)d_in[4];
    const float* Who  = (const float*)d_in[5];   // [128,1024]
    const float* bho  = (const float*)d_in[6];
    float* out = (float*)d_out;

    // workspace carve (all 256B aligned); total ~296 MB
    unsigned char* ws = (unsigned char*)d_ws;
    size_t off = 0;
    auto carve = [&](size_t bytes) {
        unsigned char* p = ws + off;
        off = (off + bytes + 255) & ~(size_t)255;
        return p;
    };
    unsigned short* xbf    = (unsigned short*)carve((size_t)BT * 256 * 2);
    unsigned short* wihbf  = (unsigned short*)carve((size_t)G4 * 256 * 2);
    unsigned short* whhbf  = (unsigned short*)carve((size_t)G4 * HH * 2);
    float*          bias   = (float*)carve((size_t)G4 * 4);
    unsigned short* xproj  = (unsigned short*)carve((size_t)BT * G4 * 2);
    unsigned short* hbuf   = (unsigned short*)carve((size_t)2 * 64 * HH * 2);
    unsigned*       cnt    = (unsigned*)carve(256);

    hipMemsetAsync(hbuf, 0, (size_t)2 * 64 * HH * 2, stream);   // h0 = 0
    hipMemsetAsync(cnt, 0, 256, stream);                        // barrier counter

    {   // converts
        int n = BT * 256;
        k_f32_to_bf16<<<(n + 255) / 256, 256, 0, stream>>>(x, xbf, n);
        n = G4 * 256;
        k_f32_to_bf16<<<(n + 255) / 256, 256, 0, stream>>>(Wih, wihbf, n);
        n = G4 * HH;
        k_f32_to_bf16<<<(n + 255) / 256, 256, 0, stream>>>(Whh, whhbf, n);
        k_bias<<<G4 / 256, 256, 0, stream>>>(bih, bhh, bias);
    }

    // x_proj GEMM
    k_xproj<<<dim3(G4 / 16, BT / 512), 256, 0, stream>>>(xbf, wihbf, bias, xproj);

    // persistent recurrence (raise dynamic-LDS cap to ~274 KB)
    hipFuncSetAttribute((const void*)k_lstm,
                        hipFuncAttributeMaxDynamicSharedMemorySize, SMEM_BYTES);
    k_lstm<<<NWG, 128, SMEM_BYTES, stream>>>(whhbf, xproj, hbuf, cnt);

    // final h lives in hbuf[0] after 512 steps
    k_final<<<(64 * 128 + 255) / 256, 256, 0, stream>>>(hbuf, Who, bho, out);
}
